// Conv6_SparseFirst_23270132809934
// MI455X (gfx1250) — compile-verified, matrix operand order checked by reading, not run on hardware
//
#include <hip/hip_runtime.h>
#include <hip/hip_bf16.h>

// ---------------------------------------------------------------------------
// CDNA5 (gfx1250) fused SparseCoding + Conv6 forward.
// All matrix math through v_wmma_f32_16x16x32_bf16 (fp32 accumulate).
// ---------------------------------------------------------------------------

typedef __attribute__((ext_vector_type(16))) __bf16 v16bf;
typedef __attribute__((ext_vector_type(8)))  float  v8f;

#define WMMA_BF16(A, B, C) \
  __builtin_amdgcn_wmma_f32_16x16x32_bf16(false, (A), false, (B), (short)0, (C), false, false)

static __device__ __forceinline__ unsigned short f2bf(float f) {
  union { float f; unsigned int u; } c; c.f = f;
  unsigned int r = c.u + 0x7FFFu + ((c.u >> 16) & 1u);
  return (unsigned short)(r >> 16);
}

// Load 16 bf16 per lane as two 16-byte chunks: k-offsets [0..7] and [16..23]
// relative to p (caller pre-offsets by half*8), matching the 16-bit A/B
// operand layout in CDNA5 ISA.
static __device__ __forceinline__ v16bf load_frag(const unsigned short* p) {
  union { uint4 q[2]; v16bf v; } t;
  t.q[0] = *(const uint4*)(p);
  t.q[1] = *(const uint4*)(p + 16);
  return t.v;
}

// ---------------------------------------------------------------------------
// Weight convert + zero-pad: fp32 [M][K] -> bf16 [Mp][Kp]
// ---------------------------------------------------------------------------
__global__ void cvt_pad_bf16(const float* __restrict__ src, unsigned short* __restrict__ dst,
                             int M, int K, int Mp, int Kp) {
  int idx = blockIdx.x * 256 + threadIdx.x;
  if (idx >= Mp * Kp) return;
  int m = idx / Kp, k = idx - m * Kp;
  float v = (m < M && k < K) ? src[(size_t)m * K + k] : 0.0f;
  dst[idx] = f2bf(v);
}

// ---------------------------------------------------------------------------
// BatchNorm training-mode statistics -> fused per-channel scale/shift.
// One block per channel. x layout [B][C][HW].
// ---------------------------------------------------------------------------
__global__ __launch_bounds__(256)
void bn_stats(const float* __restrict__ x, const float* __restrict__ g,
              const float* __restrict__ b, float* __restrict__ scale,
              float* __restrict__ shift, int C, int HW, int Bn) {
  const int c = blockIdx.x, tid = threadIdx.x;
  __shared__ float ssum[256], ssq[256];
  float s = 0.f, s2 = 0.f;
  const int total = Bn * HW;
  for (int e = tid; e < total; e += 256) {
    int n = e / HW, i = e - n * HW;
    float v = x[((size_t)n * C + c) * HW + i];
    s += v; s2 += v * v;
  }
  ssum[tid] = s; ssq[tid] = s2; __syncthreads();
  for (int off = 128; off; off >>= 1) {
    if (tid < off) { ssum[tid] += ssum[tid + off]; ssq[tid] += ssq[tid + off]; }
    __syncthreads();
  }
  if (tid == 0) {
    float inv = 1.0f / (float)total;
    float m = ssum[0] * inv;
    float var = ssq[0] * inv - m * m;
    float sc = g[c] * rsqrtf(var + 1e-5f);
    scale[c] = sc;
    shift[c] = b[c] - m * sc;
  }
}

// ---------------------------------------------------------------------------
// Fused: enc conv(3->256, 3x3) + bias + per-pixel top-k(|.|) mask,
// then dec(256->27 1x1, sigmoid, aux MSE vs im2col(x)) and red(256->64 1x1).
// Sparse code tile (256 ch x 16 px) lives only in LDS -> 236 MB tensor never
// touches HBM. 512 threads = 16 wave32.
// ---------------------------------------------------------------------------
__global__ __launch_bounds__(512)
void enc_topk_dec_red(const float* __restrict__ x,
                      const unsigned short* __restrict__ encw, const float* __restrict__ encb,
                      const unsigned short* __restrict__ decw, const float* __restrict__ decb,
                      const unsigned short* __restrict__ redw, const float* __restrict__ redb,
                      float* __restrict__ h0, float* __restrict__ aux,
                      const int* __restrict__ kptr) {
  __shared__ __attribute__((aligned(16))) unsigned short Xs[16][40];    // im2col(x), K pad 32
  __shared__ float Sv[256][17];                                         // enc output tile
  __shared__ __attribute__((aligned(16))) unsigned short Sbf[16][264];  // masked s, bf16
  __shared__ float thr[16];

  const int tid = threadIdx.x;
  const int wave = tid >> 5, lane = tid & 31;
  const int half = lane >> 4, lidx = lane & 15;
  const int pbase = blockIdx.x * 16;

  // -- phase 1: stage im2col patch of x for 16 pixels (k = ci*9+kh*3+kw, pad 27->32)
  {
    int px = tid >> 5, kk = tid & 31;
    int p = pbase + px;
    int b = p / 900; int r2 = p - b * 900;
    int oh = r2 / 30, ow = r2 - oh * 30;
    float v = 0.f;
    if (kk < 27) {
      int ci = kk / 9, rem = kk - ci * 9;
      int fr = rem / 3, fc = rem - fr * 3;
      v = x[((size_t)(b * 3 + ci) * 32 + (oh + fr)) * 32 + (ow + fc)];
    }
    Xs[px][kk] = f2bf(v);
  }
  __syncthreads();

  // -- phase 2: enc GEMM, one 16(cout)x16(px) tile per wave, K=32 (one WMMA)
  {
    const int mrow0 = wave * 16;
    v8f acc = {0.f,0.f,0.f,0.f,0.f,0.f,0.f,0.f};
    v16bf a  = load_frag(encw + (size_t)(mrow0 + lidx) * 32 + half * 8);
    v16bf bf = load_frag(&Xs[lidx][half * 8]);
    acc = WMMA_BF16(a, bf, acc);
#pragma unroll
    for (int r = 0; r < 8; ++r) {
      int co = mrow0 + half * 8 + r;
      Sv[co][lidx] = acc[r] + encb[co];
    }
  }
  __syncthreads();

  // -- phase 3: per-pixel k-th largest |value| (wave w handles pixel w)
  {
    const int px = wave;
    float vals[8];
#pragma unroll
    for (int i = 0; i < 8; ++i) vals[i] = fabsf(Sv[lane + 32 * i][px]);
    const int kk = *kptr;
    float kth = 0.f;
    for (int it = 0; it < kk; ++it) {
      float lm = -1.f; int li = 0;
#pragma unroll
      for (int i = 0; i < 8; ++i) if (vals[i] > lm) { lm = vals[i]; li = i; }
      float wm = lm;
      for (int off = 16; off; off >>= 1) wm = fmaxf(wm, __shfl_xor(wm, off, 32));
      kth = wm;
      unsigned long long ball = __ballot(lm == wm);
      int src = (int)__builtin_ctzll(ball);  // remove exactly one instance
      if (lane == src) vals[li] = -1.f;
    }
    if (lane == 0) thr[px] = kth;
  }
  __syncthreads();

  // -- phase 4: apply mask, store s tile as bf16 [pixel][channel]
  for (int e = tid; e < 4096; e += 512) {
    int co = e >> 4, px = e & 15;
    float v = Sv[co][px];
    Sbf[px][co] = f2bf((fabsf(v) >= thr[px]) ? v : 0.f);
  }
  __syncthreads();

  // -- phase 5: dec (waves 0-1, M=32 pad of 27) and red (waves 2-5, M=64)
  if (wave < 2) {
    const int mrow0 = wave * 16;
    v8f acc = {0.f,0.f,0.f,0.f,0.f,0.f,0.f,0.f};
    for (int kt = 0; kt < 8; ++kt) {
      v16bf a  = load_frag(decw + (size_t)(mrow0 + lidx) * 256 + kt * 32 + half * 8);
      v16bf bf = load_frag(&Sbf[lidx][kt * 32 + half * 8]);
      acc = WMMA_BF16(a, bf, acc);
    }
    int p = pbase + lidx;
    int b = p / 900; int r2 = p - b * 900;
    int oh = r2 / 30, ow = r2 - oh * 30;
    float lsum = 0.f;
#pragma unroll
    for (int r = 0; r < 8; ++r) {
      int m = mrow0 + half * 8 + r;
      if (m < 27) {
        float d = 1.f / (1.f + __expf(-(acc[r] + decb[m])));
        int ci = m / 9, rem = m - ci * 9;
        int fr = rem / 3, fc = rem - fr * 3;
        float t = x[((size_t)(b * 3 + ci) * 32 + (oh + fr)) * 32 + (ow + fc)];
        float df = t - d;
        lsum += df * df;
      }
    }
    for (int off = 16; off; off >>= 1) lsum += __shfl_xor(lsum, off, 32);
    if (lane == 0) atomicAdd(aux, lsum);
  } else if (wave < 6) {
    const int mrow0 = (wave - 2) * 16;
    v8f acc = {0.f,0.f,0.f,0.f,0.f,0.f,0.f,0.f};
    for (int kt = 0; kt < 8; ++kt) {
      v16bf a  = load_frag(redw + (size_t)(mrow0 + lidx) * 256 + kt * 32 + half * 8);
      v16bf bf = load_frag(&Sbf[lidx][kt * 32 + half * 8]);
      acc = WMMA_BF16(a, bf, acc);
    }
    int p = pbase + lidx;
    int b = p / 900; int r2 = p - b * 900;
    int oh = r2 / 30, ow = r2 - oh * 30;
#pragma unroll
    for (int r = 0; r < 8; ++r) {
      int co = mrow0 + half * 8 + r;
      h0[((size_t)(b * 64 + co) * 30 + oh) * 30 + ow] = acc[r] + redb[co];
    }
  }
}

// ---------------------------------------------------------------------------
// Implicit-GEMM 3x3 valid conv with fused BN(scale/shift)+ReLU on the input
// gather. Block = 8 waves = 32 cout x 64 pixels; K = Cin*9 (multiple of 32).
// ---------------------------------------------------------------------------
__global__ __launch_bounds__(256)
void conv3x3_bn_relu_wmma(const float* __restrict__ in,
                          const unsigned short* __restrict__ wbf,
                          const float* __restrict__ bias,
                          const float* __restrict__ scale,
                          const float* __restrict__ shift,
                          float* __restrict__ out,
                          int Cin, int H, int W, int Cout, int Ho, int Wo) {
  __shared__ __attribute__((aligned(16))) unsigned short Bs[64][40];  // [pixel][k], 80B stride
  const int tid  = threadIdx.x;
  const int wave = tid >> 5, lane = tid & 31;
  const int half = lane >> 4, lidx = lane & 15;
  const int mi = wave >> 2, ni = wave & 3;
  const int mrow0 = blockIdx.y * 32 + mi * 16;
  const int pbase = blockIdx.x * 64;
  const int K = Cin * 9;
  const int HoWo = Ho * Wo;

  v8f acc = {0.f,0.f,0.f,0.f,0.f,0.f,0.f,0.f};
  const unsigned short* wrow = wbf + (size_t)(mrow0 + lidx) * K;

  for (int k0 = 0; k0 < K; k0 += 32) {
    // stage B panel: 64 pixels x 32 k, BN+ReLU fused, bf16
    for (int q = tid; q < 512; q += 256) {
      int px = q >> 3;
      int k4 = (q & 7) << 2;
      int p  = pbase + px;
      int b  = p / HoWo; int r2 = p - b * HoWo;
      int oh = r2 / Wo, ow = r2 - oh * Wo;
#pragma unroll
      for (int j = 0; j < 4; ++j) {
        int kk = k0 + k4 + j;
        int ci = kk / 9, rem = kk - ci * 9;
        int fr = rem / 3, fc = rem - fr * 3;
        float v = in[((size_t)(b * Cin + ci) * H + (oh + fr)) * W + (ow + fc)];
        v = fmaxf(fmaf(v, scale[ci], shift[ci]), 0.0f);
        Bs[px][k4 + j] = f2bf(v);
      }
    }
    __syncthreads();
    v16bf a = load_frag(wrow + k0 + half * 8);
    __builtin_prefetch(wrow + k0 + 32, 0, 1);  // global_prefetch_b8 on next K-slab
    v16bf bf = load_frag(&Bs[ni * 16 + lidx][half * 8]);
    acc = WMMA_BF16(a, bf, acc);
    __syncthreads();
  }

  int p = pbase + ni * 16 + lidx;
  int b = p / HoWo; int r2 = p - b * HoWo;
  int oh = r2 / Wo, ow = r2 - oh * Wo;
#pragma unroll
  for (int r = 0; r < 8; ++r) {
    int m = mrow0 + half * 8 + r;
    out[((size_t)(b * Cout + m) * Ho + oh) * Wo + ow] = acc[r] + bias[m];
  }
}

// ---------------------------------------------------------------------------
// 2x2 max pool, stride 2.
// ---------------------------------------------------------------------------
__global__ void maxpool2k(const float* __restrict__ in, float* __restrict__ out,
                          int C, int H, int W, int Bn) {
  const int Ho = H >> 1, Wo = W >> 1;
  const int total = Bn * C * Ho * Wo;
  int idx = blockIdx.x * 256 + threadIdx.x;
  if (idx >= total) return;
  int ow = idx % Wo; int t = idx / Wo;
  int oh = t % Ho; t /= Ho;
  int c = t % C; int n = t / C;
  const float* p = in + ((size_t)(n * C + c) * H + oh * 2) * W + ow * 2;
  out[idx] = fmaxf(fmaxf(p[0], p[1]), fmaxf(p[W], p[W + 1]));
}

// ---------------------------------------------------------------------------
// Head: apply bnc scale/shift to detached features [B,256,2,2] (flatten is
// identity in this layout: j = c*4 + h*2 + w), convert to bf16.
// ---------------------------------------------------------------------------
__global__ void head_prep(const float* __restrict__ feat, const float* __restrict__ scale,
                          const float* __restrict__ shift, unsigned short* __restrict__ fbf) {
  int idx = blockIdx.x * 256 + threadIdx.x;
  if (idx >= 256 * 1024) return;
  int c = (idx & 1023) >> 2;
  fbf[idx] = f2bf(fmaf(feat[idx], scale[c], shift[c]));
}

// ---------------------------------------------------------------------------
// FC: logits[256,100] = fbf[256,1024] @ fcw[112(pad of 100),1024]^T + b.
// 112 wave-tiles (7 M x 16 N), 8 waves/block, K-loop of 32 WMMAs; operands
// are row-contiguous in k so fragments load straight from global (b128 x2).
// ---------------------------------------------------------------------------
__global__ __launch_bounds__(256)
void fc_head_wmma(const unsigned short* __restrict__ fbf,
                  const unsigned short* __restrict__ wbf,
                  const float* __restrict__ fcb,
                  float* __restrict__ logits) {
  const int tid = threadIdx.x;
  const int wave = tid >> 5, lane = tid & 31;
  const int half = lane >> 4, lidx = lane & 15;
  const int g = blockIdx.x * 8 + wave;  // 0..111
  const int mi = g >> 4, ni = g & 15;
  const int mrow0 = mi * 16, nb0 = ni * 16;

  v8f acc = {0.f,0.f,0.f,0.f,0.f,0.f,0.f,0.f};
  const unsigned short* wrow = wbf + (size_t)(mrow0 + lidx) * 1024;
  const unsigned short* frow = fbf + (size_t)(nb0 + lidx) * 1024;
  for (int k0 = 0; k0 < 1024; k0 += 32) {
    v16bf a  = load_frag(wrow + k0 + half * 8);
    __builtin_prefetch(wrow + k0 + 32, 0, 1);
    v16bf bf = load_frag(frow + k0 + half * 8);
    acc = WMMA_BF16(a, bf, acc);
  }
  const int batch = nb0 + lidx;
#pragma unroll
  for (int r = 0; r < 8; ++r) {
    int o = mrow0 + half * 8 + r;
    if (o < 100) logits[batch * 100 + o] = acc[r] + fcb[o];
  }
}

// ---------------------------------------------------------------------------
// argmax over 100 logits per batch row; one wave per row.
// ---------------------------------------------------------------------------
__global__ void argmax_k(const float* __restrict__ logits, float* __restrict__ preds) {
  const int b = blockIdx.x, lane = threadIdx.x;
  float best = -3.0e38f; int bi = 0;
  for (int o = lane; o < 100; o += 32) {
    float v = logits[b * 100 + o];
    if (v > best) { best = v; bi = o; }
  }
  for (int off = 16; off; off >>= 1) {
    float ov = __shfl_xor(best, off, 32);
    int   oi = __shfl_xor(bi, off, 32);
    if (ov > best || (ov == best && oi < bi)) { best = ov; bi = oi; }
  }
  if (lane == 0) preds[b] = (float)bi;
}

__global__ void aux_final(const float* __restrict__ aux, float* __restrict__ out) {
  out[0] = aux[0] * (1.0f / (256.0f * 27.0f * 900.0f));
}

// ---------------------------------------------------------------------------
// Workspace layout (bytes). Two ping-pong activation buffers (max 77 MB each
// -> L2-resident backbone) + small bf16 weight mirrors + BN scale/shift + aux.
// ---------------------------------------------------------------------------
static constexpr size_t SZ_BUF   = 84ull << 20;              // 88,080,384 each
static constexpr size_t OFF_A    = 0;
static constexpr size_t OFF_B    = SZ_BUF;
static constexpr size_t OFF_ENCW = 2 * SZ_BUF;               // 256x32  bf16
static constexpr size_t OFF_DECW = OFF_ENCW + 16384;         // 32x256  bf16
static constexpr size_t OFF_REDW = OFF_DECW + 16384;         // 64x256  bf16
static constexpr size_t OFF_W1   = OFF_REDW + 32768;         // 96x576
static constexpr size_t OFF_W2   = OFF_W1 + 110592;          // 128x864
static constexpr size_t OFF_W3   = OFF_W2 + 221184;          // 160x1152
static constexpr size_t OFF_W4   = OFF_W3 + 368640;          // 192x1440
static constexpr size_t OFF_W5   = OFF_W4 + 552960;          // 256x1728
static constexpr size_t OFF_FCW  = OFF_W5 + 884736;          // 112x1024
static constexpr size_t OFF_FBF  = OFF_FCW + 229376;         // 256x1024 bf16
static constexpr size_t OFF_SS   = OFF_FBF + 524288;         // scale[256]+shift[256]
static constexpr size_t OFF_AUX  = OFF_SS + 2048;            // 1 float

extern "C" void kernel_launch(void* const* d_in, const int* in_sizes, int n_in,
                              void* d_out, int out_size, void* d_ws, size_t ws_size,
                              hipStream_t stream) {
  (void)in_sizes; (void)n_in; (void)out_size; (void)ws_size;
  const float* x       = (const float*)d_in[0];
  const float* enc_w   = (const float*)d_in[1];
  const float* enc_b   = (const float*)d_in[2];
  const float* dec_w   = (const float*)d_in[3];
  const float* dec_b   = (const float*)d_in[4];
  const float* red_w   = (const float*)d_in[5];
  const float* red_b   = (const float*)d_in[6];
  const float* bn1_g   = (const float*)d_in[7];
  const float* bn1_b   = (const float*)d_in[8];
  const float* conv1_w = (const float*)d_in[9];
  const float* conv1_b = (const float*)d_in[10];
  const float* bn2_g   = (const float*)d_in[11];
  const float* bn2_b   = (const float*)d_in[12];
  const float* conv2_w = (const float*)d_in[13];
  const float* conv2_b = (const float*)d_in[14];
  const float* bn3_g   = (const float*)d_in[15];
  const float* bn3_b   = (const float*)d_in[16];
  const float* conv3_w = (const float*)d_in[17];
  const float* conv3_b = (const float*)d_in[18];
  const float* bn4_g   = (const float*)d_in[19];
  const float* bn4_b   = (const float*)d_in[20];
  const float* conv4_w = (const float*)d_in[21];
  const float* conv4_b = (const float*)d_in[22];
  const float* bn5_g   = (const float*)d_in[23];
  const float* bn5_b   = (const float*)d_in[24];
  const float* conv5_w = (const float*)d_in[25];
  const float* conv5_b = (const float*)d_in[26];
  const float* bnc_g   = (const float*)d_in[27];
  const float* bnc_b   = (const float*)d_in[28];
  const float* fc_w    = (const float*)d_in[29];
  const float* fc_b    = (const float*)d_in[30];
  const int*   kptr    = (const int*)d_in[31];

  char* ws = (char*)d_ws;
  float* bufA = (float*)(ws + OFF_A);
  float* bufB = (float*)(ws + OFF_B);
  unsigned short* encw = (unsigned short*)(ws + OFF_ENCW);
  unsigned short* decw = (unsigned short*)(ws + OFF_DECW);
  unsigned short* redw = (unsigned short*)(ws + OFF_REDW);
  unsigned short* w1   = (unsigned short*)(ws + OFF_W1);
  unsigned short* w2   = (unsigned short*)(ws + OFF_W2);
  unsigned short* w3   = (unsigned short*)(ws + OFF_W3);
  unsigned short* w4   = (unsigned short*)(ws + OFF_W4);
  unsigned short* w5   = (unsigned short*)(ws + OFF_W5);
  unsigned short* fcw  = (unsigned short*)(ws + OFF_FCW);
  unsigned short* fbf  = (unsigned short*)(ws + OFF_FBF);
  float* scale = (float*)(ws + OFF_SS);
  float* shift = scale + 256;
  float* aux   = (float*)(ws + OFF_AUX);
  float* outf  = (float*)d_out;  // [25600 logits][256 preds][1 aux]

  // --- weight mirrors (fp32 -> bf16, zero-padded) ---
  cvt_pad_bf16<<<32,   256, 0, stream>>>(enc_w,   encw, 256,   27, 256,   32);
  cvt_pad_bf16<<<32,   256, 0, stream>>>(dec_w,   decw,  27,  256,  32,  256);
  cvt_pad_bf16<<<64,   256, 0, stream>>>(red_w,   redw,  64,  256,  64,  256);
  cvt_pad_bf16<<<216,  256, 0, stream>>>(conv1_w, w1,    96,  576,  96,  576);
  cvt_pad_bf16<<<432,  256, 0, stream>>>(conv2_w, w2,   128,  864, 128,  864);
  cvt_pad_bf16<<<720,  256, 0, stream>>>(conv3_w, w3,   160, 1152, 160, 1152);
  cvt_pad_bf16<<<1080, 256, 0, stream>>>(conv4_w, w4,   192, 1440, 192, 1440);
  cvt_pad_bf16<<<1728, 256, 0, stream>>>(conv5_w, w5,   256, 1728, 256, 1728);
  cvt_pad_bf16<<<448,  256, 0, stream>>>(fc_w,    fcw,  100, 1024, 112, 1024);

  hipMemsetAsync(aux, 0, sizeof(float), stream);

  // --- fused sparse-coding front end -> h0 [256,64,30,30] in bufA ---
  enc_topk_dec_red<<<14400, 512, 0, stream>>>(x, encw, enc_b, decw, dec_b,
                                              redw, red_b, bufA, aux, kptr);

  // --- backbone: BN stats -> conv(BN+ReLU fused) -> (pool) ---
  bn_stats<<<64, 256, 0, stream>>>(bufA, bn1_g, bn1_b, scale, shift, 64, 900, 256);
  conv3x3_bn_relu_wmma<<<dim3(3136, 3), 256, 0, stream>>>(bufA, w1, conv1_b, scale, shift,
                                                          bufB, 64, 30, 30, 96, 28, 28);
  maxpool2k<<<(256*96*14*14 + 255)/256, 256, 0, stream>>>(bufB, bufA, 96, 28, 28, 256);

  bn_stats<<<96, 256, 0, stream>>>(bufA, bn2_g, bn2_b, scale, shift, 96, 196, 256);
  conv3x3_bn_relu_wmma<<<dim3(576, 4), 256, 0, stream>>>(bufA, w2, conv2_b, scale, shift,
                                                         bufB, 96, 14, 14, 128, 12, 12);

  bn_stats<<<128, 256, 0, stream>>>(bufB, bn3_g, bn3_b, scale, shift, 128, 144, 256);
  conv3x3_bn_relu_wmma<<<dim3(400, 5), 256, 0, stream>>>(bufB, w3, conv3_b, scale, shift,
                                                         bufA, 128, 12, 12, 160, 10, 10);

  bn_stats<<<160, 256, 0, stream>>>(bufA, bn4_g, bn4_b, scale, shift, 160, 100, 256);
  conv3x3_bn_relu_wmma<<<dim3(256, 6), 256, 0, stream>>>(bufA, w4, conv4_b, scale, shift,
                                                         bufB, 160, 10, 10, 192, 8, 8);
  maxpool2k<<<(256*192*4*4 + 255)/256, 256, 0, stream>>>(bufB, bufA, 192, 8, 8, 256);

  bn_stats<<<192, 256, 0, stream>>>(bufA, bn5_g, bn5_b, scale, shift, 192, 16, 256);
  conv3x3_bn_relu_wmma<<<dim3(16, 8), 256, 0, stream>>>(bufA, w5, conv5_b, scale, shift,
                                                        bufB, 192, 4, 4, 256, 2, 2);

  // --- head: bnc stats, bf16 features, FC WMMA, argmax, aux finalize ---
  bn_stats<<<256, 256, 0, stream>>>(bufB, bnc_g, bnc_b, scale, shift, 256, 4, 256);
  head_prep<<<1024, 256, 0, stream>>>(bufB, scale, shift, fbf);
  fc_head_wmma<<<14, 256, 0, stream>>>(fbf, fcw, fc_b, outf);
  argmax_k<<<256, 32, 0, stream>>>(outf, outf + 25600);
  aux_final<<<1, 1, 0, stream>>>(aux, outf + 25856);
}
